// _SharedEncoder_48455821033941
// MI455X (gfx1250) — matrix operand order repaired and checked by reading, not compiled, not run on hardware
//
#include <hip/hip_runtime.h>
#include <hip/hip_bf16.h>

typedef __attribute__((ext_vector_type(16))) _Float16 v16h;
typedef __attribute__((ext_vector_type(8)))  _Float16 v8h;
typedef __attribute__((ext_vector_type(8)))  float    v8f;

#define N_NODES   10000
#define N_EDGES   160000
#define NBATCH    64
#define NODE_DIM  64
#define EDGE_DIM  16
#define EDGE_PAD  32       // EDGE_DIM zero-padded to one WMMA K-step
#define HID       256
#define NHD       8        // heads
#define HDD       64       // head dim
#define FOUT      512      // NHD*HDD
#define NEG_SLOPE 0.2f

// ---------------------------------------------------------------- utilities
__global__ void k_zero_f32(float* __restrict__ p, int n) {
  int i = blockIdx.x * blockDim.x + threadIdx.x;
  if (i < n) p[i] = 0.0f;
}

__global__ void k_cvt_f16(const float* __restrict__ in, _Float16* __restrict__ out, int n) {
  int i = blockIdx.x * blockDim.x + threadIdx.x;
  if (i < n) out[i] = (_Float16)in[i];
}

// edge_attr [E,16] f32 -> [E,32] f16 zero-padded (K padded to 32 for WMMA)
__global__ void k_cvt_pad_edge(const float* __restrict__ in, _Float16* __restrict__ out) {
  int i = blockIdx.x * blockDim.x + threadIdx.x;
  if (i >= N_EDGES * EDGE_PAD) return;
  int e = i >> 5, k = i & 31;
  out[i] = (k < EDGE_DIM) ? (_Float16)in[e * EDGE_DIM + k] : (_Float16)0.0f;
}

// ------------------------------------------------- weight pre-swizzle (f32 -> f16)
// Wp layout = B-fragment order: tile t = kt*(F/16)+tf holds 512 f16:
//   Wp[t*512 + lane*16 + e] = W[kt*32 + ((lane&16)?8:0) + (e&7) + ((e&8)?16:0), tf*16 + (lane&15)]
// K zero-padded up to multiple of 32.
__global__ void k_prep_w(const float* __restrict__ W, _Float16* __restrict__ Wp,
                         int K, int F) {
  const int Kp = (K + 31) & ~31;
  const int total = Kp * F;
  int i = blockIdx.x * blockDim.x + threadIdx.x;
  if (i >= total) return;
  const int tile = i >> 9, within = i & 511;
  const int lane = within >> 4, e = within & 15;
  const int tilesF = F >> 4;
  const int kt = tile / tilesF, tf = tile - kt * tilesF;
  const int k = (kt << 5) + ((lane & 16) ? 8 : 0) + (e & 7) + ((e & 8) ? 16 : 0);
  const int n = (tf << 4) + (lane & 15);
  Wp[i] = (k < K) ? (_Float16)W[(size_t)k * F + n] : (_Float16)0.0f;
}

// ---------------------------------------------------------------- WMMA GEMM
// C[M,F] = A_f16[M,K] @ W + bias ; requires M%16==0, K%32==0 (pre-padded).
// One wave32 per 16x(16*FT) tile; A fragment = two 16B contiguous loads,
// B fragments = one coalesced 32B load per lane from pre-swizzled Wp.
template <int FT>
__global__ __launch_bounds__(128)
void k_gemm_wmma(const _Float16* __restrict__ A, const _Float16* __restrict__ Wp,
                 const float* __restrict__ bias,
                 _Float16* __restrict__ o16, float* __restrict__ o32,
                 int M, int K, int F) {
  const int wave    = blockIdx.x * (blockDim.x >> 5) + (threadIdx.x >> 5);
  const int lane    = threadIdx.x & 31;
  const int tilesF  = F >> 4;
  const int groupsF = tilesF / FT;
  const int waves   = (M >> 4) * groupsF;
  if (wave >= waves) return;
  const int tm   = wave / groupsF;
  const int tf0  = (wave - tm * groupsF) * FT;
  const int row0 = tm << 4;

  const int mA  = row0 + (lane & 15);
  const int kbA = (lane & 16) ? 8 : 0;
  const _Float16* Arow = A + (size_t)mA * K + kbA;
  const _Float16* Wl   = Wp + (size_t)lane * 16;

  v8f acc[FT];
#pragma unroll
  for (int t = 0; t < FT; ++t) acc[t] = (v8f){};

  const int KT = K >> 5;
  for (int kt = 0; kt < KT; ++kt) {
    const v8h alo = *(const v8h*)(Arow + (kt << 5));
    const v8h ahi = *(const v8h*)(Arow + (kt << 5) + 16);
    const v16h a = __builtin_shufflevector(alo, ahi,
        0, 1, 2, 3, 4, 5, 6, 7, 8, 9, 10, 11, 12, 13, 14, 15);
#pragma unroll
    for (int t = 0; t < FT; ++t) {
      const v16h b = *(const v16h*)(Wl + (size_t)(kt * tilesF + tf0 + t) * 512);
      acc[t] = __builtin_amdgcn_wmma_f32_16x16x32_f16(
          /*neg_a=*/false, a, /*neg_b=*/false, b,
          /*c_mod=*/(short)0, acc[t], /*reuse_a=*/false, /*reuse_b=*/false);
    }
  }

#pragma unroll
  for (int t = 0; t < FT; ++t) {
    const int nB = ((tf0 + t) << 4) + (lane & 15);
    const float bs = bias[nB];
#pragma unroll
    for (int r = 0; r < 8; ++r) {
      const int m = row0 + r + ((lane & 16) ? 8 : 0);
      const float v = acc[t][r] + bs;
      if (o16) o16[(size_t)m * F + nB] = (_Float16)v;
      if (o32) o32[(size_t)m * F + nB] = v;
    }
  }
}

// ------------------------------------------------- per-edge attention logits
// m = leaky_relu(xl[src] + xr[dst] + (ea @ ew + eb));  logit = <m, att[h]>
// segment-max via monotonic-uint atomicMax on (dst, head).
__global__ __launch_bounds__(256)
void k_edge_logits(const int* __restrict__ ei, const _Float16* __restrict__ ea,
                   const _Float16* __restrict__ xl, const _Float16* __restrict__ xr,
                   const float* __restrict__ ew, const float* __restrict__ eb,
                   const float* __restrict__ att,
                   float* __restrict__ logits, unsigned* __restrict__ mxu) {
  __shared__ float s_ew[EDGE_DIM * FOUT];   // 32 KB
  __shared__ float s_eb[FOUT];              //  2 KB
  __shared__ float s_att[FOUT];             //  2 KB
  for (int i = threadIdx.x; i < EDGE_DIM * FOUT; i += blockDim.x) s_ew[i] = ew[i];
  for (int i = threadIdx.x; i < FOUT; i += blockDim.x) { s_eb[i] = eb[i]; s_att[i] = att[i]; }
  __syncthreads();

  const int e = blockIdx.x * blockDim.x + threadIdx.x;
  if (e >= N_EDGES) return;
  const int src = ei[e];
  const int dst = ei[N_EDGES + e];

  const v16h eav16 = *(const v16h*)(ea + (size_t)e * EDGE_DIM);
  float eav[EDGE_DIM];
#pragma unroll
  for (int k = 0; k < EDGE_DIM; ++k) eav[k] = (float)eav16[k];

  const _Float16* xls = xl + (size_t)src * FOUT;
  const _Float16* xrd = xr + (size_t)dst * FOUT;

#pragma unroll
  for (int h = 0; h < NHD; ++h) {
    float logit = 0.0f;
#pragma unroll
    for (int d0 = 0; d0 < HDD; d0 += 8) {
      const v8h xlv = *(const v8h*)(xls + h * HDD + d0);
      const v8h xrv = *(const v8h*)(xrd + h * HDD + d0);
#pragma unroll
      for (int j = 0; j < 8; ++j) {
        const int f = h * HDD + d0 + j;
        float em = s_eb[f];
#pragma unroll
        for (int k = 0; k < EDGE_DIM; ++k) em += eav[k] * s_ew[k * FOUT + f];
        float m = (float)xlv[j] + (float)xrv[j] + em;
        m = (m > 0.0f) ? m : NEG_SLOPE * m;
        logit += m * s_att[f];
      }
    }
    logits[(size_t)e * NHD + h] = logit;
    const unsigned b = __float_as_uint(logit);
    const unsigned u = (b & 0x80000000u) ? ~b : (b | 0x80000000u);  // order-preserving
    atomicMax(&mxu[dst * NHD + h], u);
  }
}

// ------------------------------------------------- exp(logit - max) + denom
__global__ __launch_bounds__(256)
void k_edge_exp(const int* __restrict__ ei, float* __restrict__ logits,
                const unsigned* __restrict__ mxu, float* __restrict__ den) {
  const int p = blockIdx.x * blockDim.x + threadIdx.x;
  if (p >= N_EDGES * NHD) return;
  const int e = p >> 3, h = p & 7;
  const int dst = ei[N_EDGES + e];
  const unsigned u = mxu[dst * NHD + h];
  const unsigned bits = (u & 0x80000000u) ? (u ^ 0x80000000u) : ~u;
  const float mx = __uint_as_float(bits);
  const float ex = __expf(logits[p] - mx);
  logits[p] = ex;                       // overwrite logit with exp value
  atomicAdd(&den[dst * NHD + h], ex);
}

// ------------------------------------------------- alpha-weighted scatter-add
__global__ __launch_bounds__(256)
void k_edge_scatter(const int* __restrict__ ei, const float* __restrict__ exv,
                    const float* __restrict__ den, const _Float16* __restrict__ xl,
                    float* __restrict__ outacc) {
  const int p = blockIdx.x * blockDim.x + threadIdx.x;
  if (p >= N_EDGES * NHD) return;
  const int e = p >> 3, h = p & 7;
  const int src = ei[e];
  const int dst = ei[N_EDGES + e];
  const float alpha = exv[p] / (den[dst * NHD + h] + 1e-16f);
  const _Float16* xls = xl + (size_t)src * FOUT + h * HDD;
  float* op = outacc + (size_t)dst * FOUT + h * HDD;
#pragma unroll
  for (int d0 = 0; d0 < HDD; d0 += 8) {
    const v8h xv = *(const v8h*)(xls + d0);
#pragma unroll
    for (int j = 0; j < 8; ++j) atomicAdd(&op[d0 + j], (float)xv[j] * alpha);
  }
}

// ------------------------------------------------- finalize per-layer output
__global__ __launch_bounds__(256)
void k_finalize_concat(const float* __restrict__ outacc, const float* __restrict__ bias,
                       _Float16* __restrict__ hnext) {
  const int i = blockIdx.x * blockDim.x + threadIdx.x;
  if (i >= N_NODES * FOUT) return;
  const int f = i & (FOUT - 1);
  hnext[i] = (_Float16)(outacc[i] + bias[f]);
}

__global__ __launch_bounds__(256)
void k_finalize_mean(const float* __restrict__ outacc, const float* __restrict__ bias,
                     _Float16* __restrict__ hnext) {
  const int i = blockIdx.x * blockDim.x + threadIdx.x;
  if (i >= N_NODES * HDD) return;
  const int n = i / HDD, d = i - n * HDD;
  float s = 0.0f;
#pragma unroll
  for (int h = 0; h < NHD; ++h) s += outacc[(size_t)n * FOUT + h * HDD + d];
  hnext[i] = (_Float16)(s * (1.0f / NHD) + bias[d]);
}

// ------------------------------------------------- batch mean pooling
__global__ __launch_bounds__(256)
void k_pool(const float* __restrict__ hpost, const int* __restrict__ batch,
            float* __restrict__ pooled, float* __restrict__ cnt) {
  const int i = blockIdx.x * blockDim.x + threadIdx.x;
  if (i >= N_NODES * HID) return;
  const int n = i / HID, f = i - n * HID;
  const int b = batch[n];
  atomicAdd(&pooled[b * HID + f], hpost[i]);
  if (f == 0) atomicAdd(&cnt[b], 1.0f);
}

__global__ __launch_bounds__(256)
void k_pool_div(const float* __restrict__ pooled, const float* __restrict__ cnt,
                float* __restrict__ out) {
  const int i = blockIdx.x * blockDim.x + threadIdx.x;
  if (i >= NBATCH * HID) return;
  const int b = i / HID;
  out[i] = pooled[i] / fmaxf(cnt[b], 1.0f);
}

// ---------------------------------------------------------------- launcher
static inline void launch_gemm(const _Float16* A, const _Float16* Wp, const float* bias,
                               _Float16* o16, float* o32, int M, int K, int F,
                               hipStream_t s) {
  if ((F & 31) == 0) {
    const int waves = (M / 16) * (F / 32);
    k_gemm_wmma<2><<<(waves + 3) / 4, 128, 0, s>>>(A, Wp, bias, o16, o32, M, K, F);
  } else {
    const int waves = (M / 16) * (F / 16);
    k_gemm_wmma<1><<<(waves + 3) / 4, 128, 0, s>>>(A, Wp, bias, o16, o32, M, K, F);
  }
}

extern "C" void kernel_launch(void* const* d_in, const int* in_sizes, int n_in,
                              void* d_out, int out_size, void* d_ws, size_t ws_size,
                              hipStream_t stream) {
  (void)in_sizes; (void)n_in; (void)out_size; (void)ws_size;
  // setup_inputs() dict insertion order, params flattened in definition order.
  const float* x         = (const float*)d_in[0];
  const int*   ei        = (const int*)  d_in[1];   // [2, E]
  const float* edge_attr = (const float*)d_in[2];
  const int*   batch     = (const int*)  d_in[3];
  const float* inp_w     = (const float*)d_in[4];
  const float* inp_b     = (const float*)d_in[5];
  const float* edge_w    = (const float*)d_in[6];
  const float* edge_b    = (const float*)d_in[7];
  struct Layer { const float *lw,*lb,*rw,*rb,*ew,*eb,*att,*bias; } L[4];
  for (int i = 0; i < 4; ++i) {
    const int b = 8 + i * 8;
    L[i].lw   = (const float*)d_in[b + 0];
    L[i].lb   = (const float*)d_in[b + 1];
    L[i].rw   = (const float*)d_in[b + 2];
    L[i].rb   = (const float*)d_in[b + 3];
    L[i].ew   = (const float*)d_in[b + 4];
    L[i].eb   = (const float*)d_in[b + 5];
    L[i].att  = (const float*)d_in[b + 6];
    L[i].bias = (const float*)d_in[b + 7];
  }
  const float* post_w = (const float*)d_in[40];
  const float* post_b = (const float*)d_in[41];

  // workspace carve-out (256B aligned), ~98 MB total
  char* wp = (char*)d_ws;
  auto alloc = [&](size_t bytes) -> char* {
    char* p = wp; wp += (bytes + 255) & ~(size_t)255; return p;
  };
  _Float16* hA     = (_Float16*)alloc((size_t)N_NODES * FOUT * 2);
  _Float16* hB     = (_Float16*)alloc((size_t)N_NODES * FOUT * 2);
  _Float16* xlb    = (_Float16*)alloc((size_t)N_NODES * FOUT * 2);
  _Float16* xrb    = (_Float16*)alloc((size_t)N_NODES * FOUT * 2);
  _Float16* x16    = (_Float16*)alloc((size_t)N_NODES * NODE_DIM * 2);
  _Float16* eaIn   = (_Float16*)alloc((size_t)N_EDGES * EDGE_PAD * 2);
  _Float16* ea16   = (_Float16*)alloc((size_t)N_EDGES * EDGE_DIM * 2);
  float*    logits = (float*)   alloc((size_t)N_EDGES * NHD * 4);
  unsigned* mxu    = (unsigned*)alloc((size_t)N_NODES * NHD * 4);
  float*    den    = (float*)   alloc((size_t)N_NODES * NHD * 4);
  float*    outacc = (float*)   alloc((size_t)N_NODES * FOUT * 4);
  float*    hpost  = (float*)   alloc((size_t)N_NODES * HID * 4);
  float*    pooled = (float*)   alloc((size_t)NBATCH * HID * 4);
  float*    cnt    = (float*)   alloc((size_t)NBATCH * 4);
  // pre-swizzled f16 weights (K padded to 32)
  _Float16* wpInp  = (_Float16*)alloc((size_t)NODE_DIM * HID * 2);
  _Float16* wpEdge = (_Float16*)alloc((size_t)EDGE_PAD * EDGE_DIM * 2);
  _Float16* wpPost = (_Float16*)alloc((size_t)NODE_DIM * HID * 2);
  _Float16 *wpL[4], *wpR[4];
  for (int i = 0; i < 4; ++i) {
    const int Kin = (i == 0) ? HID : FOUT;
    wpL[i] = (_Float16*)alloc((size_t)Kin * FOUT * 2);
    wpR[i] = (_Float16*)alloc((size_t)Kin * FOUT * 2);
  }

  const int T = 256;
  auto blk = [](int n, int t) { return (n + t - 1) / t; };

  // weight pre-swizzle
  k_prep_w<<<blk(NODE_DIM * HID, T), T, 0, stream>>>(inp_w, wpInp, NODE_DIM, HID);
  k_prep_w<<<blk(EDGE_PAD * EDGE_DIM, T), T, 0, stream>>>(edge_w, wpEdge, EDGE_DIM, EDGE_DIM);
  k_prep_w<<<blk(NODE_DIM * HID, T), T, 0, stream>>>(post_w, wpPost, HDD, HID);
  for (int i = 0; i < 4; ++i) {
    const int Kin = (i == 0) ? HID : FOUT;
    k_prep_w<<<blk(Kin * FOUT, T), T, 0, stream>>>(L[i].lw, wpL[i], Kin, FOUT);
    k_prep_w<<<blk(Kin * FOUT, T), T, 0, stream>>>(L[i].rw, wpR[i], Kin, FOUT);
  }

  // input / edge projections (f16 activations, f32-accum WMMA)
  k_cvt_f16<<<blk(N_NODES * NODE_DIM, T), T, 0, stream>>>(x, x16, N_NODES * NODE_DIM);
  launch_gemm(x16, wpInp, inp_b, hA, nullptr, N_NODES, NODE_DIM, HID, stream);
  k_cvt_pad_edge<<<blk(N_EDGES * EDGE_PAD, T), T, 0, stream>>>(edge_attr, eaIn);
  launch_gemm(eaIn, wpEdge, edge_b, ea16, nullptr, N_EDGES, EDGE_PAD, EDGE_DIM, stream);

  _Float16* hcur = hA; _Float16* hnxt = hB;
  int Fin = HID;
  for (int i = 0; i < 4; ++i) {
    launch_gemm(hcur, wpL[i], L[i].lb, xlb, nullptr, N_NODES, Fin, FOUT, stream);
    launch_gemm(hcur, wpR[i], L[i].rb, xrb, nullptr, N_NODES, Fin, FOUT, stream);
    k_zero_f32<<<blk(N_NODES * NHD, T), T, 0, stream>>>((float*)mxu, N_NODES * NHD);
    k_zero_f32<<<blk(N_NODES * NHD, T), T, 0, stream>>>(den, N_NODES * NHD);
    k_zero_f32<<<blk(N_NODES * FOUT, T), T, 0, stream>>>(outacc, N_NODES * FOUT);
    k_edge_logits<<<blk(N_EDGES, T), T, 0, stream>>>(ei, ea16, xlb, xrb,
                                                     L[i].ew, L[i].eb, L[i].att,
                                                     logits, mxu);
    k_edge_exp<<<blk(N_EDGES * NHD, T), T, 0, stream>>>(ei, logits, mxu, den);
    k_edge_scatter<<<blk(N_EDGES * NHD, T), T, 0, stream>>>(ei, logits, den, xlb, outacc);
    if (i < 3) {
      k_finalize_concat<<<blk(N_NODES * FOUT, T), T, 0, stream>>>(outacc, L[i].bias, hnxt);
      Fin = FOUT;
    } else {
      k_finalize_mean<<<blk(N_NODES * HDD, T), T, 0, stream>>>(outacc, L[i].bias, hnxt);
    }
    _Float16* t = hcur; hcur = hnxt; hnxt = t;
  }

  // post projection (f32 out) + batch mean pool
  launch_gemm(hcur, wpPost, post_b, nullptr, hpost, N_NODES, HDD, HID, stream);
  k_zero_f32<<<blk(NBATCH * HID, T), T, 0, stream>>>(pooled, NBATCH * HID);
  k_zero_f32<<<blk(NBATCH, T), T, 0, stream>>>(cnt, NBATCH);
  k_pool<<<blk(N_NODES * HID, T), T, 0, stream>>>(hpost, batch, pooled, cnt);
  k_pool_div<<<blk(NBATCH * HID, T), T, 0, stream>>>(pooled, cnt, (float*)d_out);
}
